// GAT_LP_2499670966370
// MI455X (gfx1250) — matrix operand order repaired and checked by reading, not compiled, not run on hardware
//
#include <hip/hip_runtime.h>
#include <hip/hip_bf16.h>

// ---------------------------------------------------------------------------
// GAT link-prediction pipeline for gfx1250 (MI455X).
// Dominant cost: sigmoid(h @ h.T) (10000x10000x256 = 51.2 GFLOP, compute
// bound; h fits in L2) -> v_wmma_f32_16x16x32_f16 with TDM (tensor_load_to_lds)
// staged A tiles and fully vectorized b128 fragment loads. Node GEMMs use the
// same NT WMMA path with pre-transposed f16 weights. Edge softmax/aggregation
// via atomics.
// ---------------------------------------------------------------------------

typedef __attribute__((ext_vector_type(16))) _Float16 v16h;
typedef __attribute__((ext_vector_type(8)))  _Float16 v8h;
typedef __attribute__((ext_vector_type(4)))  _Float16 v4h;
typedef __attribute__((ext_vector_type(8)))  float    v8f;
typedef __attribute__((ext_vector_type(4)))  unsigned u32x4;
typedef __attribute__((ext_vector_type(8)))  int      i32x8;
typedef __attribute__((ext_vector_type(4)))  int      i32x4;

#define HEADS 4
#define HID   64
#define FDIM  256   // HEADS*HID == IN_DIM == 256

#define HAVE_TDM __has_builtin(__builtin_amdgcn_tensor_load_to_lds)

// ---------------- monotonic float <-> uint for atomic float max -------------
__device__ __forceinline__ unsigned f2mono(float f) {
  unsigned u = __float_as_uint(f);
  return (u >> 31) ? ~u : (u | 0x80000000u);
}
__device__ __forceinline__ float mono2f(unsigned m) {
  return (m >> 31) ? __uint_as_float(m ^ 0x80000000u) : __uint_as_float(~m);
}

// ---------------- WMMA fragment helpers (wave32 layouts, ISA 7.12.2) --------
// A: 16x32 f16. lane L: M = L&15 ; K-octet base = (L>>4)*8.
// halves [0..7] = K kb..kb+7 (contiguous), [8..15] = K 16+kb..16+kb+7.
__device__ __forceinline__ v16h load_a_frag(const _Float16* __restrict__ A,
                                            int lda, int m0, int k0, int lane) {
  int m  = m0 + (lane & 15);
  int kb = (lane >> 4) * 8;
  const _Float16* row = A + (size_t)m * lda + k0 + kb;
  v8h lo = *(const v8h*)row;          // 16B vector load
  v8h hi = *(const v8h*)(row + 16);   // 16B vector load
  return __builtin_shufflevector(lo, hi, 0, 1, 2, 3, 4, 5, 6, 7,
                                 8, 9, 10, 11, 12, 13, 14, 15);
}

// Same A fragment but sourced from an LDS tile As[16][lda] -> ds_load_b128.
__device__ __forceinline__ v16h load_a_frag_lds(const _Float16* As, int lda,
                                                int k0, int lane) {
  int m  = lane & 15;
  int kb = (lane >> 4) * 8;
  const _Float16* row = As + m * lda + k0 + kb;
  v8h lo = *(const v8h*)row;
  v8h hi = *(const v8h*)(row + 16);
  return __builtin_shufflevector(lo, hi, 0, 1, 2, 3, 4, 5, 6, 7,
                                 8, 9, 10, 11, 12, 13, 14, 15);
}

// B for NT GEMM: logical B[k][n] = Bt[n][k], Bt row-major [Nc x K].
// Per lane: K kb..kb+7 and 16+kb..16+kb+7 (kb=(L>>4)*8), two b128 loads.
__device__ __forceinline__ v16h load_b_frag_nt(const _Float16* __restrict__ Bt,
                                               int ldb, int k0, int n0, int lane) {
  int n  = n0 + (lane & 15);
  int kb = (lane >> 4) * 8;
  const _Float16* row = Bt + (size_t)n * ldb + k0 + kb;
  v8h lo = *(const v8h*)row;
  v8h hi = *(const v8h*)(row + 16);
  return __builtin_shufflevector(lo, hi, 0, 1, 2, 3, 4, 5, 6, 7,
                                 8, 9, 10, 11, 12, 13, 14, 15);
}

// ---------------- utility kernels -------------------------------------------
// float4 -> 4 packed halves per thread.
__global__ void k_cvt_f16x4(const float* __restrict__ s, _Float16* __restrict__ d,
                            int n4) {
  int t = blockIdx.x * blockDim.x + threadIdx.x;
  if (t >= n4) return;
  float4 v = ((const float4*)s)[t];
  v4h o = { (_Float16)v.x, (_Float16)v.y, (_Float16)v.z, (_Float16)v.w };
  ((v4h*)d)[t] = o;
}

// convert + transpose weights: Wt[n][k] = (f16)W[k][n]   (K x Nc row-major in)
__global__ void k_cvt_tr_f16(const float* __restrict__ W, _Float16* __restrict__ Wt,
                             int K, int Nc) {
  int t = blockIdx.x * blockDim.x + threadIdx.x;
  if (t >= K * Nc) return;
  int k = t / Nc, n = t - k * Nc;
  Wt[(size_t)n * K + k] = (_Float16)W[t];
}

__global__ void k_zero_u32(unsigned* __restrict__ p, int n) {
  int t = blockIdx.x * blockDim.x + threadIdx.x;
  if (t < n) p[t] = 0u;
}

// -------- WMMA GEMM NT (direct): C[M,Nc] = A[M,K] * Bt[Nc,K]^T --------------
// One wave computes a 16x64 strip (A fragment reused across 4 WMMAs).
__global__ void k_gemm_nt(const _Float16* __restrict__ A,
                          const _Float16* __restrict__ Bt,
                          float* __restrict__ C, int M, int Nc, int K) {
  int wave = (blockIdx.x * blockDim.x + threadIdx.x) >> 5;
  int lane = threadIdx.x & 31;
  int strips = Nc >> 6;                       // Nc multiple of 64 here
  int total  = (M >> 4) * strips;
  if (wave >= total) return;                  // wave-uniform
  int tm = wave / strips, ts = wave % strips;
  int m0 = tm << 4, nbase = ts << 6;
  v8f acc[4] = {{}, {}, {}, {}};
  for (int k0 = 0; k0 < K; k0 += 32) {
    v16h a = load_a_frag(A, K, m0, k0, lane);
#pragma unroll
    for (int s = 0; s < 4; ++s) {
      v16h b = load_b_frag_nt(Bt, K, k0, nbase + (s << 4), lane);
      acc[s] = __builtin_amdgcn_wmma_f32_16x16x32_f16(false, a, false, b,
                                                      (short)0, acc[s],
                                                      false, false);
    }
  }
#pragma unroll
  for (int s = 0; s < 4; ++s) {
    int n = nbase + (s << 4) + (lane & 15);
    int rb = (lane >> 4) * 8;
#pragma unroll
    for (int r = 0; r < 8; ++r)
      C[(size_t)(m0 + rb + r) * Nc + n] = acc[s][r];
  }
}

// ------- adj = sigmoid(H @ H^T): TDM-staged A tile, 8 strips per block ------
// Block: 8 waves share one 16x256 A tile in LDS (loaded by the Tensor Data
// Mover, tracked on TENSORcnt); wave w handles one 64-col strip.
__global__ void k_adj_sigmoid(const _Float16* __restrict__ Hh,
                              float* __restrict__ adj, int N) {
  __shared__ _Float16 As[16 * FDIM];          // 8 KB
  const int K = FDIM;
  int strips  = (N + 63) >> 6;
  int sblocks = (strips + 7) >> 3;
  int tm = blockIdx.x / sblocks;
  int sb = blockIdx.x - tm * sblocks;
  int wave = threadIdx.x >> 5, lane = threadIdx.x & 31;
  int m0 = tm << 4;

#if HAVE_TDM
  if (threadIdx.x < 32) {                     // one wave issues the DMA
    unsigned long long ga = (unsigned long long)(uintptr_t)(Hh + (size_t)m0 * K);
    unsigned la = (unsigned)(uintptr_t)As;    // LDS byte offset
    // D# group 0: count=1 | lds_addr | global_addr[56:0] | type=2
    u32x4 g0 = { 1u, la, (unsigned)ga,
                 (unsigned)((ga >> 32) & 0x01FFFFFFu) | (2u << 30) };
    // D# group 1: data_size=2B; tensor_dim0=256; tensor_dim1=16;
    //             tile_dim0=256; tile_dim1=16; tensor_dim0_stride=256
    i32x8 g1 = { (int)(1u << 16),             // [17:16] data_size=1 (2 bytes)
                 (int)(256u << 16),           // tensor_dim0[15:0] @ [79:64]... [63:48]
                 (int)(16u << 16),            // tensor_dim0 hi=0 | tensor_dim1 lo=16
                 (int)(256u << 16),           // tensor_dim1 hi=0 | tile_dim0=256
                 16,                          // tile_dim1=16, tile_dim2=0
                 256,                         // tensor_dim0_stride lo
                 0, 0 };
    // groups 2/3: higher dims set to 1 (inactive), strides/tiles 0
    i32x4 g2 = { 1, 1, 0, 0 };
    i32x4 g3 = { 0, 1 << 16, 0, 0 };
#if defined(__clang_major__) && __clang_major__ >= 23
    i32x8 g4 = { 0, 0, 0, 0, 0, 0, 0, 0 };
    __builtin_amdgcn_tensor_load_to_lds(g0, g1, g2, g3, g4, 0);
#else
    __builtin_amdgcn_tensor_load_to_lds(g0, g1, g2, g3, 0);
#endif
    __builtin_amdgcn_s_wait_tensorcnt(0);
  }
#else
  {
    int idx = threadIdx.x * 16;               // half index in tile
    int r = idx >> 8, c = idx & 255;
    *(v16h*)(As + idx) = *(const v16h*)(Hh + (size_t)(m0 + r) * K + c);
  }
#endif
  __syncthreads();

  int strip = sb * 8 + wave;
  if (strip >= strips) return;                // wave-uniform, after the sync
  int nbase = strip << 6;

  // prefetch first B rows of this strip into cache hierarchy
  __builtin_prefetch(Hh + (size_t)(nbase + (lane & 15)) * K, 0, 3);

  v8f acc[4] = {{}, {}, {}, {}};
  for (int k0 = 0; k0 < K; k0 += 32) {
    v16h a = load_a_frag_lds(As, K, k0, lane);
#pragma unroll
    for (int s = 0; s < 4; ++s) {
      int n0 = nbase + (s << 4);
      if (n0 < N) {                           // uniform per wave (N%16==0)
        v16h b = load_b_frag_nt(Hh, K, k0, n0, lane);
        acc[s] = __builtin_amdgcn_wmma_f32_16x16x32_f16(false, a, false, b,
                                                        (short)0, acc[s],
                                                        false, false);
      }
    }
  }
#pragma unroll
  for (int s = 0; s < 4; ++s) {
    int n0 = nbase + (s << 4);
    if (n0 < N) {
      int n = n0 + (lane & 15), rb = (lane >> 4) * 8;
#pragma unroll
      for (int r = 0; r < 8; ++r) {
        float x = acc[s][r];
        adj[(size_t)(m0 + rb + r) * N + n] = 1.0f / (1.0f + __expf(-x));
      }
    }
  }
}

// ---------------- attention scalar products el / er -------------------------
__global__ void k_elr(const float* __restrict__ ft, const float* __restrict__ al,
                      const float* __restrict__ ar, float* __restrict__ el,
                      float* __restrict__ er, int N) {
  int t = blockIdx.x * blockDim.x + threadIdx.x;
  if (t >= N * HEADS) return;
  int n = t >> 2, h = t & 3;
  const float* f = ft + (size_t)n * FDIM + h * HID;
  const float* a = al + h * HID;
  const float* b = ar + h * HID;
  float sl = 0.f, sr = 0.f;
#pragma unroll 8
  for (int d = 0; d < HID; ++d) { sl += f[d] * a[d]; sr += f[d] * b[d]; }
  el[t] = sl; er[t] = sr;
}

// ---------------- edge phase: segment max / exp-sum / aggregate -------------
__global__ void k_edge_max(const int* __restrict__ src, const int* __restrict__ dst,
                           const float* __restrict__ el, const float* __restrict__ er,
                           unsigned* __restrict__ emax_u, int E) {
  int t = blockIdx.x * blockDim.x + threadIdx.x;
  if (t >= E * HEADS) return;
  int e = t >> 2, h = t & 3;
  float v = el[src[e] * HEADS + h] + er[dst[e] * HEADS + h];
  v = v > 0.f ? v : 0.2f * v;                 // leaky_relu
  atomicMax(&emax_u[dst[e] * HEADS + h], f2mono(v));
}

__global__ void k_edge_exp(const int* __restrict__ src, const int* __restrict__ dst,
                           const float* __restrict__ el, const float* __restrict__ er,
                           const unsigned* __restrict__ emax_u,
                           float* __restrict__ alpha, float* __restrict__ denom, int E) {
  int t = blockIdx.x * blockDim.x + threadIdx.x;
  if (t >= E * HEADS) return;
  int e = t >> 2, h = t & 3;
  int d = dst[e];
  float v = el[src[e] * HEADS + h] + er[d * HEADS + h];
  v = v > 0.f ? v : 0.2f * v;
  float ee = __expf(v - mono2f(emax_u[d * HEADS + h]));
  alpha[t] = ee;
  atomicAdd(&denom[d * HEADS + h], ee);
}

// One wave per edge; each lane handles 8 consecutive features (same head).
__global__ void k_edge_agg(const int* __restrict__ src, const int* __restrict__ dst,
                           const float* __restrict__ alpha, const float* __restrict__ denom,
                           const float* __restrict__ ft, float* __restrict__ rst, int E) {
  long long t = (long long)blockIdx.x * blockDim.x + threadIdx.x;
  if (t >= (long long)E * 32) return;
  int e = (int)(t >> 5), lane = (int)(t & 31);
  int s = src[e], d = dst[e];
  int h = lane >> 3;                          // 8 feats * 8 lanes = 64 per head
  float a = alpha[e * HEADS + h] / denom[d * HEADS + h];
  int j0 = lane * 8;
  const float4* f = (const float4*)(ft + (size_t)s * FDIM + j0);
  float*        r = rst + (size_t)d * FDIM + j0;
  float4 f0 = f[0], f1 = f[1];
  atomicAdd(&r[0], f0.x * a); atomicAdd(&r[1], f0.y * a);
  atomicAdd(&r[2], f0.z * a); atomicAdd(&r[3], f0.w * a);
  atomicAdd(&r[4], f1.x * a); atomicAdd(&r[5], f1.y * a);
  atomicAdd(&r[6], f1.z * a); atomicAdd(&r[7], f1.w * a);
}

// ---------------- bias + residual + ELU, dual-precision output --------------
__global__ void k_finalize(const float* __restrict__ rst, const float* __restrict__ resid,
                           const float* __restrict__ bias, float* __restrict__ of32,
                           _Float16* __restrict__ of16, int N) {
  int t = blockIdx.x * blockDim.x + threadIdx.x;
  if (t >= N * FDIM) return;
  int j = t & (FDIM - 1);
  float x = rst[t] + bias[j];
  if (resid) x += resid[t];
  float y = x > 0.f ? x : (__expf(x) - 1.0f);  // elu
  of32[t] = y;
  of16[t] = (_Float16)y;
}

// ---------------- seq_fts = inputs @ Wfc (fp32, small) ----------------------
__global__ void k_seq(const float* __restrict__ X, const float* __restrict__ W,
                      float* __restrict__ out, int N) {
  int t = blockIdx.x * blockDim.x + threadIdx.x;
  if (t >= N * HID) return;
  int n = t >> 6, d = t & 63;
  const float* x = X + (size_t)n * FDIM;
  float acc = 0.f;
#pragma unroll 8
  for (int k = 0; k < FDIM; ++k) acc += x[k] * W[k * HID + d];
  out[t] = acc;
}

// ---------------------------------------------------------------------------
static inline size_t alignup(size_t x) { return (x + 255) & ~(size_t)255; }

extern "C" void kernel_launch(void* const* d_in, const int* in_sizes, int n_in,
                              void* d_out, int out_size, void* d_ws, size_t ws_size,
                              hipStream_t stream) {
  const float* inputs = (const float*)d_in[0];
  const int*   src    = (const int*)d_in[1];
  const int*   dst    = (const int*)d_in[2];
  const float* W0     = (const float*)d_in[3];
  const float* al0    = (const float*)d_in[4];
  const float* ar0    = (const float*)d_in[5];
  const float* b0     = (const float*)d_in[6];
  const float* W1     = (const float*)d_in[7];
  const float* al1    = (const float*)d_in[8];
  const float* ar1    = (const float*)d_in[9];
  const float* b1     = (const float*)d_in[10];
  const float* Wfc    = (const float*)d_in[11];

  const int N = in_sizes[0] / FDIM;     // 10000
  const int E = in_sizes[1];            // 320000

  float* adj  = (float*)d_out;
  float* hout = adj + (size_t)N * N;
  float* seq  = hout + (size_t)N * FDIM;

  // ---- carve workspace ----
  char* p = (char*)d_ws;
  _Float16* Xh    = (_Float16*)p; p += alignup((size_t)N * FDIM * 2);
  _Float16* W0t   = (_Float16*)p; p += alignup((size_t)FDIM * FDIM * 2);
  _Float16* W1t   = (_Float16*)p; p += alignup((size_t)FDIM * FDIM * 2);
  float*    ft    = (float*)p;    p += alignup((size_t)N * FDIM * 4);
  float*    el    = (float*)p;    p += alignup((size_t)N * HEADS * 4);
  float*    er    = (float*)p;    p += alignup((size_t)N * HEADS * 4);
  unsigned* emaxu = (unsigned*)p; p += alignup((size_t)N * HEADS * 4);
  float*    denom = (float*)p;    p += alignup((size_t)N * HEADS * 4);
  float*    alpha = (float*)p;    p += alignup((size_t)E * HEADS * 4);
  float*    rst   = (float*)p;    p += alignup((size_t)N * FDIM * 4);
  float*    h0    = (float*)p;    p += alignup((size_t)N * FDIM * 4);
  _Float16* h0h   = (_Float16*)p; p += alignup((size_t)N * FDIM * 2);
  _Float16* hh    = (_Float16*)p; p += alignup((size_t)N * FDIM * 2);
  (void)ws_size; (void)n_in; (void)out_size;

  const int B = 256;
  auto blocks = [](long long work, int b) { return (int)((work + b - 1) / b); };

  // ---- conversions (vectorized) + weight transposes ----
  k_cvt_f16x4<<<blocks((long long)N * FDIM / 4, B), B, 0, stream>>>(inputs, Xh, N * FDIM / 4);
  k_cvt_tr_f16<<<blocks(FDIM * FDIM, B), B, 0, stream>>>(W0, W0t, FDIM, FDIM);
  k_cvt_tr_f16<<<blocks(FDIM * FDIM, B), B, 0, stream>>>(W1, W1t, FDIM, FDIM);

  const int gemmWaves  = (N / 16) * (FDIM / 64);
  const int gemmBlocks = blocks((long long)gemmWaves * 32, B);

  // ================= GAT layer 0 =================
  k_gemm_nt<<<gemmBlocks, B, 0, stream>>>(Xh, W0t, ft, N, FDIM, FDIM);
  k_elr<<<blocks((long long)N * HEADS, B), B, 0, stream>>>(ft, al0, ar0, el, er, N);
  k_zero_u32<<<blocks((long long)N * HEADS, B), B, 0, stream>>>(emaxu, N * HEADS);
  k_zero_u32<<<blocks((long long)N * HEADS, B), B, 0, stream>>>((unsigned*)denom, N * HEADS);
  k_zero_u32<<<blocks((long long)N * FDIM, B), B, 0, stream>>>((unsigned*)rst, N * FDIM);
  k_edge_max<<<blocks((long long)E * HEADS, B), B, 0, stream>>>(src, dst, el, er, emaxu, E);
  k_edge_exp<<<blocks((long long)E * HEADS, B), B, 0, stream>>>(src, dst, el, er, emaxu, alpha, denom, E);
  k_edge_agg<<<blocks((long long)E * 32, B), B, 0, stream>>>(src, dst, alpha, denom, ft, rst, E);
  k_finalize<<<blocks((long long)N * FDIM, B), B, 0, stream>>>(rst, nullptr, b0, h0, h0h, N);

  // ================= GAT layer 1 (residual) =================
  k_gemm_nt<<<gemmBlocks, B, 0, stream>>>(h0h, W1t, ft, N, FDIM, FDIM);
  k_elr<<<blocks((long long)N * HEADS, B), B, 0, stream>>>(ft, al1, ar1, el, er, N);
  k_zero_u32<<<blocks((long long)N * HEADS, B), B, 0, stream>>>(emaxu, N * HEADS);
  k_zero_u32<<<blocks((long long)N * HEADS, B), B, 0, stream>>>((unsigned*)denom, N * HEADS);
  k_zero_u32<<<blocks((long long)N * FDIM, B), B, 0, stream>>>((unsigned*)rst, N * FDIM);
  k_edge_max<<<blocks((long long)E * HEADS, B), B, 0, stream>>>(src, dst, el, er, emaxu, E);
  k_edge_exp<<<blocks((long long)E * HEADS, B), B, 0, stream>>>(src, dst, el, er, emaxu, alpha, denom, E);
  k_edge_agg<<<blocks((long long)E * 32, B), B, 0, stream>>>(src, dst, alpha, denom, ft, rst, E);
  k_finalize<<<blocks((long long)N * FDIM, B), B, 0, stream>>>(rst, h0, b1, hout, hh, N);

  // ================= adj_rec = sigmoid(h @ h.T) — dominant GEMM =============
  {
    int strips = (N + 63) / 64;               // 157
    int sblocks = (strips + 7) / 8;           // 20
    int grid = (N / 16) * sblocks;            // 12500 blocks
    k_adj_sigmoid<<<grid, B, 0, stream>>>(hh, adj, N);
  }

  // ================= seq_fts = inputs @ Wfc (fp32) ===========================
  k_seq<<<blocks((long long)N * HID, B), B, 0, stream>>>(inputs, Wfc, seq, N);
}
